// MemoryEfficientLiTLoss_31542239822372
// MI455X (gfx1250) — matrix operand order repaired and checked by reading, not compiled, not run on hardware
//
#include <hip/hip_runtime.h>
#include <math.h>

#define NROWS 16384
#define DDIM  256
#define INV_T (1.0f / 0.07f)

typedef __attribute__((ext_vector_type(16))) __bf16 v16bf;
typedef __attribute__((ext_vector_type(8)))  __bf16 v8bf;   // 16 bytes -> b128
typedef __attribute__((ext_vector_type(8)))  float  v8f;
typedef __attribute__((ext_vector_type(4)))  float  v4f;

// ---------------------------------------------------------------------------
// Kernel 1: fp32 normalize rows of text/image, emit bf16 copies + fp32 diagonal
// One wave (32 lanes) per row; 8 elements per lane.
// ---------------------------------------------------------------------------
__global__ void __launch_bounds__(256)
lit_normalize_kernel(const float* __restrict__ text, const float* __restrict__ img,
                     __bf16* __restrict__ Xb, __bf16* __restrict__ Yb,
                     float* __restrict__ diag) {
    const int wave = threadIdx.x >> 5;
    const int lane = threadIdx.x & 31;
    const int row  = blockIdx.x * 8 + wave;

    const float* tx = text + (size_t)row * DDIM + lane * 8;
    const float* ty = img  + (size_t)row * DDIM + lane * 8;
    v4f x0 = *(const v4f*)tx;
    v4f x1 = *(const v4f*)(tx + 4);
    v4f y0 = *(const v4f*)ty;
    v4f y1 = *(const v4f*)(ty + 4);

    float ssx = 0.0f, ssy = 0.0f;
#pragma unroll
    for (int i = 0; i < 4; ++i) {
        ssx += x0[i] * x0[i] + x1[i] * x1[i];
        ssy += y0[i] * y0[i] + y1[i] * y1[i];
    }
#pragma unroll
    for (int m = 1; m < 32; m <<= 1) {
        ssx += __shfl_xor(ssx, m, 32);
        ssy += __shfl_xor(ssy, m, 32);
    }
    const float sx = 1.0f / fmaxf(sqrtf(ssx), 1e-12f);
    const float sy = 1.0f / fmaxf(sqrtf(ssy), 1e-12f);

    float dot = 0.0f;
    v8bf xb, yb;
#pragma unroll
    for (int i = 0; i < 4; ++i) {
        float xv = x0[i] * sx, yv = y0[i] * sy;
        dot += xv * yv;
        xb[i] = (__bf16)xv;
        yb[i] = (__bf16)yv;
    }
#pragma unroll
    for (int i = 0; i < 4; ++i) {
        float xv = x1[i] * sx, yv = y1[i] * sy;
        dot += xv * yv;
        xb[4 + i] = (__bf16)xv;
        yb[4 + i] = (__bf16)yv;
    }
    *(v8bf*)(Xb + (size_t)row * DDIM + lane * 8) = xb;
    *(v8bf*)(Yb + (size_t)row * DDIM + lane * 8) = yb;

#pragma unroll
    for (int m = 1; m < 32; m <<= 1) dot += __shfl_xor(dot, m, 32);
    if (lane == 0) diag[row] = dot;
}

// ---------------------------------------------------------------------------
// Fragment loaders per CDNA5 WMMA VGPR layouts (CDNA5 ISA §7.12.2).
// A (16x32 bf16): lane L -> row M=L%16, h=L/16; VGPR0-3: K=8h..8h+7,
//                 VGPR4-7: K=16+8h..16+8h+7  (two b128 loads, 16 apart)
// B (32x16 bf16): lane L -> col N=L%16, h=L/16; VGPR0-3: K=8h..8h+7,
//                 VGPR4-7: K=16+8h..16+8h+7  (two b128 loads, 16 apart)
// ---------------------------------------------------------------------------
__device__ __forceinline__ v16bf load_a_frag(const __bf16* p) {
    v8bf lo = *(const v8bf*)p;          // K = k0 + 8h .. +7
    v8bf hi = *(const v8bf*)(p + 16);   // K = k0 + 16 + 8h .. +7
    v16bf r;
#pragma unroll
    for (int i = 0; i < 8; ++i) { r[i] = lo[i]; r[8 + i] = hi[i]; }
    return r;
}

__device__ __forceinline__ v16bf load_b_frag(const __bf16* p) {
    v8bf lo = *(const v8bf*)p;          // K = k0 + 8h .. +7
    v8bf hi = *(const v8bf*)(p + 16);   // K = k0 + 16 + 8h .. +7
    v16bf r;
#pragma unroll
    for (int i = 0; i < 8; ++i) { r[i] = lo[i]; r[8 + i] = hi[i]; }
    return r;
}

// ---------------------------------------------------------------------------
// Kernel 2: per-row sum of exp((x_i . y_j - 1)/t) over all j, via bf16 WMMA.
// 128 threads = 4 waves; each wave owns a 16-row strip (A resident in VGPRs),
// streams all 1024 16-column tiles of Y (L2-resident), K=256 = 8 WMMA steps.
// ---------------------------------------------------------------------------
__global__ void __launch_bounds__(128)
lit_rowsum_kernel(const __bf16* __restrict__ Xb, const __bf16* __restrict__ Yb,
                  float* __restrict__ sumexp) {
    const int wave = threadIdx.x >> 5;
    const int lane = threadIdx.x & 31;
    const int m    = lane & 15;     // row within A tile / col within B tile
    const int h    = lane >> 4;     // half-wave selector
    const int row0 = (blockIdx.x * 4 + wave) * 16;

    // Preload the wave's entire 16x256 A strip: 8 fragments, 64 VGPRs.
    const __bf16* arow = Xb + (size_t)(row0 + m) * DDIM + h * 8;
    v16bf a[8];
#pragma unroll
    for (int k = 0; k < 8; ++k) a[k] = load_a_frag(arow + k * 32);

    v8f sums;
#pragma unroll
    for (int r = 0; r < 8; ++r) sums[r] = 0.0f;

    for (int j = 0; j < NROWS; j += 16) {
        const __bf16* bcol = Yb + (size_t)(j + m) * DDIM + h * 8;
        v8f c = {};
#pragma unroll
        for (int k = 0; k < 8; ++k) {
            v16bf b = load_b_frag(bcol + k * 32);
            c = __builtin_amdgcn_wmma_f32_16x16x32_bf16(
                    /*neg_a=*/false, a[k], /*neg_b=*/false, b,
                    /*c_mod=*/(short)0, c, /*reuse_a=*/false, /*reuse_b=*/false);
        }
        // exp((S-1)/t); v_exp_f32 is a TRANS op and co-executes with XDL WMMA
#pragma unroll
        for (int r = 0; r < 8; ++r)
            sums[r] += __expf((c[r] - 1.0f) * INV_T);
    }

    // C layout: lane L, VGPR r  <->  row (r + 8h), col (L%16).
    // Reduce across the 16 lanes of each half-wave.
#pragma unroll
    for (int mask = 1; mask < 16; mask <<= 1) {
#pragma unroll
        for (int r = 0; r < 8; ++r)
            sums[r] += __shfl_xor(sums[r], mask, 32);
    }
    if (m == 0) {
#pragma unroll
        for (int r = 0; r < 8; ++r)
            sumexp[row0 + h * 8 + r] = sums[r];
    }
}

// ---------------------------------------------------------------------------
// Kernel 3: loss = -mean( (diag-1)/t - log(sumexp) )
// ---------------------------------------------------------------------------
__global__ void __launch_bounds__(256)
lit_finalize_kernel(const float* __restrict__ diag, const float* __restrict__ sumexp,
                    float* __restrict__ out) {
    __shared__ float red[256];
    float acc = 0.0f;
    for (int i = threadIdx.x; i < NROWS; i += 256)
        acc += (diag[i] - 1.0f) * INV_T - logf(sumexp[i]);
    red[threadIdx.x] = acc;
    __syncthreads();
#pragma unroll
    for (int s = 128; s > 0; s >>= 1) {
        if (threadIdx.x < s) red[threadIdx.x] += red[threadIdx.x + s];
        __syncthreads();
    }
    if (threadIdx.x == 0) out[0] = -red[0] / (float)NROWS;
}

extern "C" void kernel_launch(void* const* d_in, const int* in_sizes, int n_in,
                              void* d_out, int out_size, void* d_ws, size_t ws_size,
                              hipStream_t stream) {
    const float* text = (const float*)d_in[0];
    const float* img  = (const float*)d_in[1];
    float* out = (float*)d_out;

    char* ws = (char*)d_ws;
    const size_t fb = (size_t)NROWS * DDIM * sizeof(__bf16);   // 8 MB per matrix
    __bf16* Xb     = (__bf16*)ws;
    __bf16* Yb     = (__bf16*)(ws + fb);
    float*  diag   = (float*)(ws + 2 * fb);
    float*  sumexp = diag + NROWS;

    lit_normalize_kernel<<<NROWS / 8, 256, 0, stream>>>(text, img, Xb, Yb, diag);
    lit_rowsum_kernel<<<NROWS / 64, 128, 0, stream>>>(Xb, Yb, sumexp);
    lit_finalize_kernel<<<1, 256, 0, stream>>>(diag, sumexp, out);
}
